// YOLOLoss_44341242364521
// MI455X (gfx1250) — compile-verified, matrix operand order checked by reading, not run on hardware
//
#include <hip/hip_runtime.h>
#include <hip/hip_bf16.h>
#include <math.h>

// ---------------------------------------------------------------------------
// YOLO loss for MI455X (gfx1250, wave32).
// 259,584 anchor cells; each wave handles 16 anchors:
//   - stage 16x85 pred floats into a private LDS region (b128 coalesced)
//   - log-softmax row reduction (80 classes) done on the matrix unit:
//       3x v_wmma_f32_16x16x32_f16 with a ones B-matrix -> row sums of exp
//   - lanes 0..15 do the scalar BCE / box-MSE / IoU terms
//   - wave shuffle-reduce 6 partials, one atomicAdd set per wave
// ---------------------------------------------------------------------------

typedef __attribute__((ext_vector_type(16))) _Float16 v16h;
typedef __attribute__((ext_vector_type(8)))  float    v8f;

#define N_ANCH   259584          // 32*52*52*3
#define GROUPS   16224           // N_ANCH / 16
#define WPB      8               // waves per block
#define NBLOCKS  (GROUPS / WPB)  // 2028

__global__ void yolo_init_ws(float* ws) {
    if (threadIdx.x < 8) ws[threadIdx.x] = 0.0f;
}

__global__ __launch_bounds__(256)
void yolo_main(const float* __restrict__ tgt,   // (N_ANCH, 6)
               const float* __restrict__ pred,  // (N_ANCH, 85) contiguous
               const float* __restrict__ anc,   // (3, 2)
               float* __restrict__ ws)          // 6 accumulators
{
    __shared__ float sh[WPB * 1360 + 32];       // 16 anchors * 85 floats / wave
    const int lane  = threadIdx.x & 31;
    const int wave  = threadIdx.x >> 5;
    const int group = blockIdx.x * WPB + wave;  // grid sized exactly: no bounds
    float* lp = &sh[wave * 1360];

    // ---- stage 16 anchors x 85 logits into LDS, 128-bit coalesced ----------
    {
        const float4* src = (const float4*)(pred + (size_t)group * 1360);
        float4* dst = (float4*)lp;
        #pragma unroll
        for (int i = 0; i < 11; ++i) {
            int idx = lane + i * 32;            // 340 float4 per group
            if (idx < 340) dst[idx] = src[idx];
        }
    }
    __syncthreads();

    // ---- per-row class max (row m split between lanes L and L^16) ----------
    const int hi = lane >> 4;                   // which K-half this lane owns
    const int m  = lane & 15;                   // matrix row / anchor in group
    const float* row = lp + m * 85 + 5;         // 80 class logits of row m

    float pmax = -3.4e38f;
    #pragma unroll
    for (int chunk = 0; chunk < 3; ++chunk) {
        #pragma unroll
        for (int j = 0; j < 16; ++j) {
            int k  = chunk * 32 + hi * 8 + (j < 8 ? j : j + 8); // ISA A layout
            int kc = (k < 80) ? k : 0;
            float v = row[kc];
            pmax = (k < 80) ? fmaxf(pmax, v) : pmax;
        }
    }
    float rmax = fmaxf(pmax, __shfl_xor(pmax, 16, 32));

    // ---- sum(exp(l - max)) via WMMA: D = E(16x96 f16) x ones(96x16) --------
    v16h ones;
    #pragma unroll
    for (int j = 0; j < 16; ++j) ones[j] = (_Float16)1.0f;

    v8f c = {};
    #pragma unroll
    for (int chunk = 0; chunk < 3; ++chunk) {
        v16h a;
        #pragma unroll
        for (int j = 0; j < 16; ++j) {
            int k  = chunk * 32 + hi * 8 + (j < 8 ? j : j + 8);
            int kc = (k < 80) ? k : 0;
            float e = __expf(row[kc] - rmax);
            a[j] = (_Float16)((k < 80) ? e : 0.0f);
        }
        c = __builtin_amdgcn_wmma_f32_16x16x32_f16(
                /*neg_a=*/false, a, /*neg_b=*/false, ones,
                /*c_mod=*/(short)0, c, /*reuse_a=*/false, /*reuse_b=*/false);
    }

    // Route sumexp(row m) to lane: lanes0-15 hold rows0-7, lanes16-31 rows8-15
    int sel = m & 7;
    float x0 = (sel & 4)
        ? ((sel & 2) ? ((sel & 1) ? c[7] : c[6]) : ((sel & 1) ? c[5] : c[4]))
        : ((sel & 2) ? ((sel & 1) ? c[3] : c[2]) : ((sel & 1) ? c[1] : c[0]));
    float xs = __shfl_xor(x0, 24, 32);
    float sumexp = ((lane >= 8) && (lane < 24)) ? xs : x0;

    // ---- scalar per-anchor terms on lanes 0..15 ----------------------------
    float acc0 = 0.f, acc1 = 0.f, acc2 = 0.f, acc3 = 0.f, acc4 = 0.f, acc5 = 0.f;
    if (lane < 16) {
        const int n = group * 16 + m;
        const float* t = tgt + (size_t)n * 6;
        float t0 = t[0], t1 = t[1], t2 = t[2], t3 = t[3], t4 = t[4], t5 = t[5];
        int aidx = n - (n / 3) * 3;
        float aw = anc[2 * aidx], ah = anc[2 * aidx + 1];
        const float* p = lp + m * 85;
        float p0 = p[0], p1 = p[1], p2 = p[2], p3 = p[3], p4 = p[4];

        float obj   = (t0 == 1.0f) ? 1.0f : 0.0f;
        float noobj = (t0 == 0.0f) ? 1.0f : 0.0f;

        // BCE-with-logits on objectness (no-object term)
        float bce = fmaxf(p0, 0.0f) - p0 * t0 + log1pf(__expf(-fabsf(p0)));

        // sparse CE from log-softmax (class term)
        int   label = (int)t5;
        float ll    = lp[m * 85 + 5 + label];
        float ce    = rmax + __logf(sumexp) - ll;

        // box MSE in regression space
        float sx  = 1.0f / (1.0f + __expf(-p1));
        float sy  = 1.0f / (1.0f + __expf(-p2));
        float twl = __logf(1e-16f + t3 / aw);
        float thl = __logf(1e-16f + t4 / ah);
        float dx = t1 - sx, dy = t2 - sy, dw = twl - p3, dh = thl - p4;
        float box = dx * dx + dy * dy + dw * dw + dh * dh;

        // objectness MSE vs midpoint IoU (decoded pred vs log-space target,
        // faithfully reproducing the reference)
        float pw = __expf(p3) * aw, ph = __expf(p4) * ah;
        float b1x1 = sx - 0.5f * pw, b1x2 = sx + 0.5f * pw;
        float b1y1 = sy - 0.5f * ph, b1y2 = sy + 0.5f * ph;
        float b2x1 = t1 - 0.5f * twl, b2x2 = t1 + 0.5f * twl;
        float b2y1 = t2 - 0.5f * thl, b2y2 = t2 + 0.5f * thl;
        float iw = fmaxf(fminf(b1x2, b2x2) - fmaxf(b1x1, b2x1), 0.0f);
        float ih = fmaxf(fminf(b1y2, b2y2) - fmaxf(b1y1, b2y1), 0.0f);
        float inter = iw * ih;
        float a1 = fabsf((b1x2 - b1x1) * (b1y2 - b1y1));
        float a2 = fabsf((b2x2 - b2x1) * (b2y2 - b2y1));
        float iou = inter / (a1 + a2 - inter + 1e-6f);
        float om  = iou * t0 - p0;

        acc0 = obj;
        acc1 = noobj;
        acc2 = bce * noobj;
        acc3 = ce  * obj;
        acc4 = box * obj;
        acc5 = om * om * obj;
    }

    // ---- wave reduce (5 shfl_xor steps) + one atomic set per wave ----------
    float av[6] = {acc0, acc1, acc2, acc3, acc4, acc5};
    #pragma unroll
    for (int q = 0; q < 6; ++q) {
        float v = av[q];
        #pragma unroll
        for (int off = 16; off >= 1; off >>= 1) v += __shfl_xor(v, off, 32);
        if (lane == 0) atomicAdd(&ws[q], v);
    }
}

__global__ void yolo_finalize(const float* __restrict__ ws, float* __restrict__ out) {
    if (threadIdx.x == 0 && blockIdx.x == 0) {
        float n_obj   = fmaxf(ws[0], 1.0f);
        float n_noobj = fmaxf(ws[1], 1.0f);
        float noobjloss = ws[2] / n_noobj;
        float classloss = ws[3] / n_obj;
        float boxloss   = ws[4] / (4.0f * n_obj);
        float objloss   = ws[5] / n_obj;
        out[0] = 10.0f * noobjloss + 1.0f * classloss
               + 10.0f * boxloss   + 1.0f * objloss;
    }
}

extern "C" void kernel_launch(void* const* d_in, const int* in_sizes, int n_in,
                              void* d_out, int out_size, void* d_ws, size_t ws_size,
                              hipStream_t stream) {
    const float* tgt  = (const float*)d_in[0];   // (B,S,S,3,6)
    const float* pred = (const float*)d_in[1];   // (B,S,S,255)
    const float* anc  = (const float*)d_in[2];   // (3,2)
    float* ws  = (float*)d_ws;
    float* out = (float*)d_out;

    yolo_init_ws<<<1, 32, 0, stream>>>(ws);
    yolo_main<<<NBLOCKS, 256, 0, stream>>>(tgt, pred, anc, ws);
    yolo_finalize<<<1, 32, 0, stream>>>(ws, out);
}
